// CrossAttentionEAF_36893769073119
// MI455X (gfx1250) — matrix operand order repaired and checked by reading, not compiled, not run on hardware
//
#include <hip/hip_runtime.h>
#include <math.h>

#define D      128
#define QTOT   10000
#define NKTOT  4224
#define EPS    1e-5f

typedef __attribute__((ext_vector_type(16))) _Float16 v16h;
typedef __attribute__((ext_vector_type(8)))  _Float16 v8h;
typedef __attribute__((ext_vector_type(8)))  float    v8f;

// ---------------- block-wide sum over 128 threads ----------------
__device__ __forceinline__ float blk_sum128(float v, float* red) {
    int t = threadIdx.x;
    red[t] = v; __syncthreads();
    #pragma unroll
    for (int s = 64; s > 0; s >>= 1) {
        if (t < s) red[t] += red[t + s];
        __syncthreads();
    }
    float r = red[0]; __syncthreads();
    return r;
}

__device__ __forceinline__ v16h cat8(v8h lo, v8h hi) {
    v16h r;
    #pragma unroll
    for (int i = 0; i < 8; ++i) { r[i] = lo[i]; r[i + 8] = hi[i]; }
    return r;
}

// ---------------- kernel 1: LN(k)@Wk -> kf (row major f16); LN(v)@Wv -> vT (transposed f16)
__global__ __launch_bounds__(128)
void proj_kv_kernel(const float* __restrict__ k, const float* __restrict__ v,
                    const float* __restrict__ kn_g, const float* __restrict__ kn_b,
                    const float* __restrict__ vn_g, const float* __restrict__ vn_b,
                    const float* __restrict__ Wk, const float* __restrict__ Wv,
                    _Float16* __restrict__ kf, _Float16* __restrict__ vT) {
    __shared__ float red[128];
    __shared__ float skx[128];
    __shared__ float svx[128];
    const int nk  = blockIdx.x;          // 0..NK-1  (n*704 + y*44 + x)
    const int d   = threadIdx.x;
    const int n   = nk / 704;
    const int rem = nk % 704;
    float kx = k[(n * D + d) * 704 + rem];
    float vx = v[(n * D + d) * 704 + rem];

    float mu  = blk_sum128(kx, red) * (1.f / 128.f);
    float dv  = kx - mu;
    float var = blk_sum128(dv * dv, red) * (1.f / 128.f);
    float kn  = dv * rsqrtf(var + EPS) * kn_g[d] + kn_b[d];

    mu  = blk_sum128(vx, red) * (1.f / 128.f);
    dv  = vx - mu;
    var = blk_sum128(dv * dv, red) * (1.f / 128.f);
    float vn = dv * rsqrtf(var + EPS) * vn_g[d] + vn_b[d];

    skx[d] = kn; svx[d] = vn; __syncthreads();

    float yk = 0.f, yv = 0.f;
    #pragma unroll 4
    for (int i = 0; i < 128; ++i) {
        yk += skx[i] * Wk[i * 128 + d];
        yv += svx[i] * Wv[i * 128 + d];
    }
    kf[nk * 128 + d]   = (_Float16)yk;
    vT[d * NKTOT + nk] = (_Float16)yv;
}

// ---------------- kernel 2: LN(q)@Wq * scale -> qf (row major f16)
__global__ __launch_bounds__(128)
void proj_q_kernel(const float* __restrict__ q,
                   const float* __restrict__ qn_g, const float* __restrict__ qn_b,
                   const float* __restrict__ Wq, _Float16* __restrict__ qf) {
    __shared__ float red[128];
    __shared__ float sx[128];
    const int qi = blockIdx.x;
    const int d  = threadIdx.x;
    float x   = q[d * QTOT + qi];
    float mu  = blk_sum128(x, red) * (1.f / 128.f);
    float dv  = x - mu;
    float var = blk_sum128(dv * dv, red) * (1.f / 128.f);
    float xn  = dv * rsqrtf(var + EPS) * qn_g[d] + qn_b[d];
    sx[d] = xn; __syncthreads();
    float y = 0.f;
    #pragma unroll 4
    for (int i = 0; i < 128; ++i) y += sx[i] * Wq[i * 128 + d];
    qf[qi * 128 + d] = (_Float16)(y * 0.17677669529663687f);   // fold 1/sqrt(32)
}

// ---------------- kernel 3: flash attention with multiplicative EAF mask (WMMA f16)
// grid = Q/16 blocks; block = 128 threads = 4 waves; wave w handles head w.
// W_logits tile [16x32] staged once into LDS and shared by all 4 heads.
__global__ __launch_bounds__(128)
void attn_kernel(const _Float16* __restrict__ qf, const _Float16* __restrict__ kf,
                 const _Float16* __restrict__ vT, const float* __restrict__ Wl,
                 float* __restrict__ attn_out) {
    __shared__ __align__(32) float    sWl[16 * 32];
    __shared__ __align__(32) _Float16 sP[4][16 * 32];

    const int qt    = blockIdx.x;
    const int head  = threadIdx.x >> 5;
    const int lane  = threadIdx.x & 31;
    const int l16   = lane & 15;
    const int hi    = lane >> 4;          // half-wave selector
    const int qbase = qt * 16;
    const int dbase = head * 32;

    // A tile: 16x32 f16 q-slice, loop invariant.
    const _Float16* arow = qf + (size_t)(qbase + l16) * D + dbase;
    const v16h aQ = cat8(*(const v8h*)(arow + hi * 16),
                         *(const v8h*)(arow + hi * 16 + 8));

    v8f acc0 = {}; v8f acc1 = {};
    float mrun[8], lrun[8];
    #pragma unroll
    for (int r = 0; r < 8; ++r) { mrun[r] = -INFINITY; lrun[r] = 0.f; }

    for (int kt = 0; kt < NKTOT; kt += 32) {
        // stage W_logits tile cooperatively (contiguous 128B rows)
        #pragma unroll
        for (int i = 0; i < 4; ++i) {
            int idx = threadIdx.x + i * 128;
            sWl[idx] = Wl[(size_t)(qbase + (idx >> 5)) * NKTOT + kt + (idx & 31)];
        }
        if (kt + 32 < NKTOT) {
            #pragma unroll
            for (int i = 0; i < 4; ++i) {
                int idx = threadIdx.x + i * 128;
                __builtin_prefetch(&Wl[(size_t)(qbase + (idx >> 5)) * NKTOT + (kt + 32) + (idx & 31)], 0, 1);
            }
        }
        __syncthreads();

        // logits: S = qA @ K^T  (two 16x16 tiles; B rows contiguous in kf)
        const v16h b0 = *(const v16h*)(kf + (size_t)(kt + l16) * D + dbase + hi * 16);
        const v16h b1 = *(const v16h*)(kf + (size_t)(kt + 16 + l16) * D + dbase + hi * 16);
        v8f zc = {};
        v8f s0 = __builtin_amdgcn_wmma_f32_16x16x32_f16(false, aQ, false, b0, (short)0, zc, false, false);
        v8f s1 = __builtin_amdgcn_wmma_f32_16x16x32_f16(false, aQ, false, b1, (short)0, zc, false, false);

        // multiplicative mask + online softmax stats
        float p0[8], p1[8], mx[8];
        #pragma unroll
        for (int r = 0; r < 8; ++r) {
            p0[r] = s0[r] * sWl[(r + hi * 8) * 32 + l16];
            p1[r] = s1[r] * sWl[(r + hi * 8) * 32 + 16 + l16];
            mx[r] = fmaxf(p0[r], p1[r]);
        }
        #pragma unroll
        for (int m = 1; m < 16; m <<= 1) {     // row-max across 16 lanes of half-wave
            #pragma unroll
            for (int r = 0; r < 8; ++r) {
                float o = __int_as_float(__builtin_amdgcn_ds_bpermute(((lane ^ m) << 2),
                                                                     __float_as_int(mx[r])));
                mx[r] = fmaxf(mx[r], o);
            }
        }
        float rs[8];
        #pragma unroll
        for (int r = 0; r < 8; ++r) {
            float mnew  = fmaxf(mrun[r], mx[r]);
            float alpha = __expf(mrun[r] - mnew);
            p0[r] = __expf(p0[r] - mnew);
            p1[r] = __expf(p1[r] - mnew);
            mrun[r]  = mnew;
            lrun[r] *= alpha;
            acc0[r] *= alpha;
            acc1[r] *= alpha;
            rs[r] = p0[r] + p1[r];
        }
        #pragma unroll
        for (int m = 1; m < 16; m <<= 1) {     // row-sum across 16 lanes
            #pragma unroll
            for (int r = 0; r < 8; ++r) {
                rs[r] += __int_as_float(__builtin_amdgcn_ds_bpermute(((lane ^ m) << 2),
                                                                    __float_as_int(rs[r])));
            }
        }
        #pragma unroll
        for (int r = 0; r < 8; ++r) lrun[r] += rs[r];

        // C-layout -> A-layout re-tile of P via per-wave LDS (DS ops in-order per wave)
        _Float16* sp = sP[head];
        #pragma unroll
        for (int r = 0; r < 8; ++r) {
            sp[(r + hi * 8) * 32 + l16]      = (_Float16)p0[r];
            sp[(r + hi * 8) * 32 + 16 + l16] = (_Float16)p1[r];
        }
        __builtin_amdgcn_wave_barrier();
        const v16h pA = cat8(*(const v8h*)(sp + l16 * 32 + hi * 16),
                             *(const v8h*)(sp + l16 * 32 + hi * 16 + 8));

        // PV: out += P @ V   (B tiles contiguous thanks to transposed vT)
        const v16h bv0 = *(const v16h*)(vT + (size_t)(dbase + l16) * NKTOT + kt + hi * 16);
        const v16h bv1 = *(const v16h*)(vT + (size_t)(dbase + 16 + l16) * NKTOT + kt + hi * 16);
        acc0 = __builtin_amdgcn_wmma_f32_16x16x32_f16(false, pA, false, bv0, (short)0, acc0, false, false);
        acc1 = __builtin_amdgcn_wmma_f32_16x16x32_f16(false, pA, false, bv1, (short)0, acc1, false, false);

        __syncthreads();   // protect sWl reuse
    }

    #pragma unroll
    for (int r = 0; r < 8; ++r) {
        float inv = 1.f / lrun[r];
        int row = qbase + r + hi * 8;
        attn_out[(size_t)row * D + dbase + l16]      = acc0[r] * inv;
        attn_out[(size_t)row * D + dbase + 16 + l16] = acc1[r] * inv;
    }
}

// ---------------- kernel 4: out-proj + skip + preLN + GELU MLP + postLN, transposed store
__global__ __launch_bounds__(128)
void mlp_kernel(const float* __restrict__ attn_out, const float* __restrict__ skip,
                const float* __restrict__ Wp, const float* __restrict__ bp,
                const float* __restrict__ pre_g, const float* __restrict__ pre_b,
                const float* __restrict__ W1, const float* __restrict__ b1,
                const float* __restrict__ W2, const float* __restrict__ b2,
                const float* __restrict__ post_g, const float* __restrict__ post_b,
                float* __restrict__ out) {
    __shared__ float red[128];
    __shared__ float sx[128];
    __shared__ float sh1[256];
    const int qi = blockIdx.x;
    const int j  = threadIdx.x;

    sx[j] = attn_out[(size_t)qi * 128 + j];
    __syncthreads();
    float z = bp[j];
    #pragma unroll 4
    for (int i = 0; i < 128; ++i) z += sx[i] * Wp[i * 128 + j];
    z += skip[(size_t)j * QTOT + qi];

    float mu  = blk_sum128(z, red) * (1.f / 128.f);
    float dv  = z - mu;
    float var = blk_sum128(dv * dv, red) * (1.f / 128.f);
    z = dv * rsqrtf(var + EPS) * pre_g[j] + pre_b[j];

    sx[j] = z;                 // safe: all prior sx reads fenced by blk_sum barriers
    __syncthreads();
    #pragma unroll
    for (int c = 0; c < 2; ++c) {
        int jj = j + c * 128;
        float a = b1[jj];
        #pragma unroll 4
        for (int i = 0; i < 128; ++i) a += sx[i] * W1[i * 256 + jj];
        sh1[jj] = 0.5f * a * (1.f + erff(a * 0.70710678118654752f));   // exact GELU
    }
    __syncthreads();
    float t = b2[j];
    #pragma unroll 4
    for (int i = 0; i < 256; ++i) t += sh1[i] * W2[i * 128 + j];
    z = z + t;

    mu  = blk_sum128(z, red) * (1.f / 128.f);
    dv  = z - mu;
    var = blk_sum128(dv * dv, red) * (1.f / 128.f);
    z = dv * rsqrtf(var + EPS) * post_g[j] + post_b[j];

    out[(size_t)j * QTOT + qi] = z;
}

extern "C" void kernel_launch(void* const* d_in, const int* in_sizes, int n_in,
                              void* d_out, int out_size, void* d_ws, size_t ws_size,
                              hipStream_t stream) {
    const float* q      = (const float*)d_in[0];
    const float* k      = (const float*)d_in[1];
    const float* v      = (const float*)d_in[2];
    const float* Wl     = (const float*)d_in[3];
    const float* skip   = (const float*)d_in[4];
    const float* qn_g   = (const float*)d_in[5];
    const float* qn_b   = (const float*)d_in[6];
    const float* kn_g   = (const float*)d_in[7];
    const float* kn_b   = (const float*)d_in[8];
    const float* vn_g   = (const float*)d_in[9];
    const float* vn_b   = (const float*)d_in[10];
    const float* Wq     = (const float*)d_in[11];
    const float* Wk     = (const float*)d_in[12];
    const float* Wv     = (const float*)d_in[13];
    const float* Wp     = (const float*)d_in[14];
    const float* bp     = (const float*)d_in[15];
    const float* pre_g  = (const float*)d_in[16];
    const float* pre_b  = (const float*)d_in[17];
    const float* W1     = (const float*)d_in[18];
    const float* b1     = (const float*)d_in[19];
    const float* W2     = (const float*)d_in[20];
    const float* b2     = (const float*)d_in[21];
    const float* post_g = (const float*)d_in[22];
    const float* post_b = (const float*)d_in[23];

    char* ws = (char*)d_ws;
    size_t off = 0;
    _Float16* qf = (_Float16*)(ws + off); off += (size_t)QTOT  * 128 * 2; off = (off + 255) & ~(size_t)255;
    _Float16* kf = (_Float16*)(ws + off); off += (size_t)NKTOT * 128 * 2; off = (off + 255) & ~(size_t)255;
    _Float16* vT = (_Float16*)(ws + off); off += (size_t)128 * NKTOT * 2; off = (off + 255) & ~(size_t)255;
    float* attn_out = (float*)(ws + off); off += (size_t)QTOT * 128 * 4;

    proj_kv_kernel<<<NKTOT, 128, 0, stream>>>(k, v, kn_g, kn_b, vn_g, vn_b, Wk, Wv, kf, vT);
    proj_q_kernel<<<QTOT, 128, 0, stream>>>(q, qn_g, qn_b, Wq, qf);
    attn_kernel<<<QTOT / 16, 128, 0, stream>>>(qf, kf, vT, Wl, attn_out);
    mlp_kernel<<<QTOT, 128, 0, stream>>>(attn_out, skip, Wp, bp, pre_g, pre_b,
                                         W1, b1, W2, b2, post_g, post_b, (float*)d_out);
}
